// ODE_16561393894089
// MI455X (gfx1250) — matrix operand order repaired and checked by reading, not compiled, not run on hardware
//
#include <hip/hip_runtime.h>
#include <cstdint>
#include <cstddef>

// ---------------------------------------------------------------------------
// Hypernetwork ODE step, fused for MI455X (gfx1250).
//   emb   = mean_d( D @ W_enc^T + b_enc )              (tiny, VALU kernel)
//   layer: h' = tanh( einsum(h, W_b) + b_b ) where W_b = emb @ W_hyp-slice
// Fused as one bf16 WMMA GEMM per layer:
//   h'[b,o] = sum_{k'=i*256+c} (h[b,i]*emb[b,c]) * W_hyp[(off+o*wi)*256 + k']
//           + sum_i h[b,i]*b_hyp[off+o*wi+i] + emb[b,:].W_hyp[boff+o,:] + b_hyp[boff+o]
// B-matrix column o is a contiguous span of W_hyp -> W_hyp streamed exactly
// once from HBM (605 MB @ 23.3 TB/s ~ 26 us lower bound). A-operand (emb x h
// outer product) is built in-registers from an LDS-staged tile so L2 serves
// only the W_hyp stream.
// ---------------------------------------------------------------------------

typedef __attribute__((ext_vector_type(16))) __bf16 v16bf;
typedef __attribute__((ext_vector_type(8)))  float  v8f;
typedef __attribute__((ext_vector_type(4)))  float  f32x4;

#define LATENT  64
#define HIDDEN  512
#define CODE    256
#define DS      5
#define GLDIM   60      // GL*DIM
#define BATCH   256
#define HSTRIDE 512     // stride of intermediate h buffers

#define EPAD    260     // emb tile LDS row stride (260%64==4 -> bank-conflict free)
#define HPADMAX 516     // h tile LDS row stride for wi=512 (wi+4)

// ---------------------------------------------------------------------------
// Encoder: emb[b,c] = (1/DS) * sum_d sum_k D[b,d,k]*W_enc[c,k] + b_enc[c]
// Also seeds h0[b, 0:64] = z[b, 0:64].
// ---------------------------------------------------------------------------
__global__ __launch_bounds__(256)
void hyper_encoder_kernel(const float* __restrict__ D,
                          const float* __restrict__ W_enc,
                          const float* __restrict__ b_enc,
                          const float* __restrict__ z,
                          float* __restrict__ emb,
                          float* __restrict__ h0)
{
    __shared__ float sD[DS * GLDIM];           // 300 floats
    const int b = blockIdx.x;
    const int c = threadIdx.x;

    for (int t = threadIdx.x; t < DS * GLDIM; t += blockDim.x)
        sD[t] = D[(size_t)b * (DS * GLDIM) + t];
    __syncthreads();

    const float* wr = W_enc + (size_t)c * GLDIM;
    float acc = 0.f;
    #pragma unroll
    for (int d = 0; d < DS; ++d) {
        float s = 0.f;
        #pragma unroll
        for (int k = 0; k < GLDIM; ++k)
            s += sD[d * GLDIM + k] * wr[k];
        acc += s;
    }
    emb[(size_t)b * CODE + c] = acc * (1.0f / DS) + b_enc[c];

    if (c < LATENT)
        h0[(size_t)b * HSTRIDE + c] = z[(size_t)b * LATENT + c];
}

// ---------------------------------------------------------------------------
// Fused hyper-layer GEMM. One wave per block; wave owns a 16b x 64o tile.
// emb/h tiles staged in LDS; W_hyp streamed from HBM with prefetch.
// ---------------------------------------------------------------------------
__global__ __launch_bounds__(32)
void hyper_layer_kernel(const float* __restrict__ W_hyp,
                        const float* __restrict__ b_hyp,
                        const float* __restrict__ emb,
                        const float* __restrict__ h_in,
                        float* __restrict__ h_out,
                        int wi, long long off, long long boff,
                        int out_stride, int apply_tanh)
{
    __shared__ __align__(16) float semb[16 * EPAD];     // 16.6 KB
    __shared__ __align__(16) float sh[16 * HPADMAX];    // 33.0 KB

    const int lane = threadIdx.x;        // 0..31
    const int half = lane >> 4;          // 0: low K half, 1: high K half
    const int col  = lane & 15;
    const int b0   = blockIdx.x * 16;
    const int o0   = blockIdx.y * 64;
    const int hpad = wi + 4;             // 68 or 516: stride mod 64 == 4

    // ---- stage emb tile (16 x 256) into LDS ----
    for (int t = lane; t < 16 * 64; t += 32) {
        const int row = t >> 6;
        const int q   = (t & 63) << 2;
        *(f32x4*)(&semb[row * EPAD + q]) =
            *(const f32x4*)(emb + (size_t)(b0 + row) * CODE + q);
    }
    // ---- stage h tile (16 x wi) into LDS ----
    #pragma unroll
    for (int row = 0; row < 16; ++row)
        for (int q = lane; q < (wi >> 2); q += 32)
            *(f32x4*)(&sh[row * hpad + (q << 2)]) =
                *(const f32x4*)(h_in + (size_t)(b0 + row) * HSTRIDE + (q << 2));
    __syncthreads();

    v8f acc[4] = {v8f{}, v8f{}, v8f{}, v8f{}};

    // A-matrix (M = h (x) emb): lane supplies row b0+col (from LDS).
    const float* embrow = semb + col * EPAD;
    const float* hrow   = sh   + col * hpad;

    // B-matrix: lane supplies column o0 + ot*16 + col (contiguous W_hyp span).
    const float* wcol[4];
    #pragma unroll
    for (int ot = 0; ot < 4; ++ot) {
        const long long o = o0 + ot * 16 + col;
        wcol[ot] = W_hyp + ((size_t)off + (size_t)o * wi) * CODE;
    }

    const long long ktot = (long long)wi * CODE;
    for (long long k0 = 0; k0 < ktot; k0 += 32) {
        const int   i  = (int)(k0 >> 8);      // 256 c-values per i, 32 | 256
        const int   c0 = (int)(k0 & 255);
        const float hv = hrow[i];

        // ---- A fragment from LDS: K in {half*16..+15} ----
        const float* ep = embrow + c0 + half * 16;
        const f32x4 e0 = *(const f32x4*)(ep);
        const f32x4 e1 = *(const f32x4*)(ep + 4);
        const f32x4 e2 = *(const f32x4*)(ep + 8);
        const f32x4 e3 = *(const f32x4*)(ep + 12);
        v16bf a;
        #pragma unroll
        for (int j = 0; j < 4; ++j) {
            a[j]      = (__bf16)(e0[j] * hv);
            a[4 + j]  = (__bf16)(e1[j] * hv);
            a[8 + j]  = (__bf16)(e2[j] * hv);
            a[12 + j] = (__bf16)(e3[j] * hv);
        }

        // ---- 4 B fragments share the A fragment ----
        #pragma unroll
        for (int ot = 0; ot < 4; ++ot) {
            const float* wp = wcol[ot] + k0 + half * 16;  // 16 consecutive K
            __builtin_prefetch((const void*)(wp + 2048), 0, 0); // ~64 steps ahead
            const f32x4 w0 = *(const f32x4*)(wp);
            const f32x4 w1 = *(const f32x4*)(wp + 4);
            const f32x4 w2 = *(const f32x4*)(wp + 8);
            const f32x4 w3 = *(const f32x4*)(wp + 12);
            v16bf bb;
            #pragma unroll
            for (int j = 0; j < 4; ++j) {
                bb[j]      = (__bf16)w0[j];
                bb[4 + j]  = (__bf16)w1[j];
                bb[8 + j]  = (__bf16)w2[j];
                bb[12 + j] = (__bf16)w3[j];
            }
            acc[ot] = __builtin_amdgcn_wmma_f32_16x16x32_bf16(
                          false, a, false, bb, (short)0, acc[ot], false, false);
        }
    }

    // ---- epilogue: hyper-generated bias + outer bias, tanh, store ----
    // C layout: reg r, lane -> row b0 + r + 8*half, col o0 + ot*16 + (lane&15)
    #pragma unroll
    for (int ot = 0; ot < 4; ++ot) {
        const int o = o0 + ot * 16 + col;
        float t[8] = {0.f, 0.f, 0.f, 0.f, 0.f, 0.f, 0.f, 0.f};

        // emb[b,:] . W_hyp[boff+o, :]   (emb rows from LDS)
        const float* wb = W_hyp + (size_t)(boff + o) * CODE;
        for (int c = 0; c < CODE; ++c) {
            const float wv = wb[c];
            #pragma unroll
            for (int r = 0; r < 8; ++r)
                t[r] += semb[(r + half * 8) * EPAD + c] * wv;
        }
        // sum_i h[b,i] * b_hyp[off + o*wi + i]   (h rows from LDS)
        const float* bh = b_hyp + (size_t)off + (size_t)o * wi;
        for (int i2 = 0; i2 < wi; ++i2) {
            const float bv = bh[i2];
            #pragma unroll
            for (int r = 0; r < 8; ++r)
                t[r] += sh[(r + half * 8) * hpad + i2] * bv;
        }
        const float bo = b_hyp[boff + o];
        #pragma unroll
        for (int r = 0; r < 8; ++r) {
            float v = acc[ot][r] + t[r] + bo;
            if (apply_tanh) v = tanhf(v);
            h_out[(size_t)(b0 + r + half * 8) * out_stride + o] = v;
        }
    }
}

// ---------------------------------------------------------------------------
extern "C" void kernel_launch(void* const* d_in, const int* in_sizes, int n_in,
                              void* d_out, int out_size, void* d_ws, size_t ws_size,
                              hipStream_t stream)
{
    (void)in_sizes; (void)n_in; (void)out_size; (void)ws_size;
    // setup_inputs order: t, z, D, W_enc, b_enc, W_hyp, b_hyp  (t unused)
    const float* z     = (const float*)d_in[1];
    const float* D     = (const float*)d_in[2];
    const float* W_enc = (const float*)d_in[3];
    const float* b_enc = (const float*)d_in[4];
    const float* W_hyp = (const float*)d_in[5];
    const float* b_hyp = (const float*)d_in[6];
    float* out = (float*)d_out;

    // workspace: emb (256KB) | hA (512KB) | hB (512KB)
    char* ws  = (char*)d_ws;
    float* emb = (float*)(ws);
    float* hA  = (float*)(ws + (size_t)BATCH * CODE * 4);
    float* hB  = (float*)(ws + (size_t)BATCH * CODE * 4 + (size_t)BATCH * HSTRIDE * 4);

    hyper_encoder_kernel<<<BATCH, 256, 0, stream>>>(D, W_enc, b_enc, z, emb, hA);

    // layer 0: (512,64)   weights @0,      bias @32768
    hyper_layer_kernel<<<dim3(BATCH / 16, HIDDEN / 64), 32, 0, stream>>>(
        W_hyp, b_hyp, emb, hA, hB, /*wi=*/LATENT, /*off=*/0LL, /*boff=*/32768LL,
        HSTRIDE, /*tanh=*/1);
    // layer 1: (512,512)  weights @33280,  bias @295424
    hyper_layer_kernel<<<dim3(BATCH / 16, HIDDEN / 64), 32, 0, stream>>>(
        W_hyp, b_hyp, emb, hB, hA, HIDDEN, 33280LL, 295424LL, HSTRIDE, 1);
    // layer 2: (512,512)  weights @295936, bias @558080
    hyper_layer_kernel<<<dim3(BATCH / 16, HIDDEN / 64), 32, 0, stream>>>(
        W_hyp, b_hyp, emb, hA, hB, HIDDEN, 295936LL, 558080LL, HSTRIDE, 1);
    // layer 3: (64,512)   weights @558592, bias @591360 -> d_out (B x 64, f32)
    hyper_layer_kernel<<<dim3(BATCH / 16, LATENT / 64), 32, 0, stream>>>(
        W_hyp, b_hyp, emb, hB, out, HIDDEN, 558592LL, 591360LL, LATENT, 0);
}
